// DotProductAttention_51024211476693
// MI455X (gfx1250) — matrix operand order repaired and checked by reading, not compiled, not run on hardware
//
#include <hip/hip_runtime.h>
#include <hip/hip_bf16.h>

// DotProductAttention with softmax over the BATCH axis (per reference).
// B=8, S=4096, D=64, fp32 in/out.
//
// Strategy (MI455X): Q/K/V/O ~= 34MB fits in the 192MB L2, so the kernel is
// matrix-core/L2 bound. K is pre-converted to f16 and V to f16 TRANSPOSED
// ([b][d][s]) in d_ws once, so the hot loop is only b128 f16 loads + v_wmma +
// the cross-batch softmax. Attn weights are staged in LDS directly in WMMA
// A-operand layout so the reload is a single b128-pair ds_load. Scheduling
// fences keep load clauses grouped ahead of the WMMA chains, and V tiles are
// issued before the softmax so the barrier/exp work hides their latency.
//
// Mapping: 1 workgroup (256 thr = 8 waves, wave32) = one 16-row q tile for
// ALL 8 batches; wave w owns batch b=w. Loop over keys in chunks of 32.

typedef __attribute__((ext_vector_type(16))) _Float16 v16h;
typedef __attribute__((ext_vector_type(8)))  float    v8f;

#define BATCH 8
#define SEQ   4096
#define DIM   64
#define QT    16   // q rows per workgroup
#define KT    32   // keys per inner chunk (== WMMA K for attn@V)

#if __has_builtin(__builtin_amdgcn_sched_barrier)
#define SCHED_FENCE() __builtin_amdgcn_sched_barrier(0)
#else
#define SCHED_FENCE()
#endif

static __device__ __forceinline__ v8f wmma_f16(v16h a, v16h b, v8f c) {
    // (neg_a, A, neg_b, B, c_mod, C, reuse_a, reuse_b)
    return __builtin_amdgcn_wmma_f32_16x16x32_f16(false, a, false, b,
                                                  (short)0, c, false, false);
}

// ---------------------------------------------------------------------------
// Preprocess 1: elementwise f32 -> f16 convert of K (coalesced, 4 elems/thr).
// ---------------------------------------------------------------------------
__global__ __launch_bounds__(256)
void convert_k_kernel(const float* __restrict__ src, _Float16* __restrict__ dst) {
    const size_t i = ((size_t)blockIdx.x * 256 + threadIdx.x) * 4;
    const float4 f = *(const float4*)(src + i);
    dst[i + 0] = (_Float16)f.x;
    dst[i + 1] = (_Float16)f.y;
    dst[i + 2] = (_Float16)f.z;
    dst[i + 3] = (_Float16)f.w;
}

// ---------------------------------------------------------------------------
// Preprocess 2: V[b][s][d] (f32) -> Vt16[b][d][s] (f16), LDS-tiled transpose.
// One block per (batch, 64-key slab): 64x64 tile.
// ---------------------------------------------------------------------------
__global__ __launch_bounds__(256)
void transpose_v_kernel(const float* __restrict__ V, _Float16* __restrict__ Vt) {
    __shared__ _Float16 tile[64][66];   // pad to break bank conflicts

    const int b  = blockIdx.x >> 6;          // batch
    const int s0 = (blockIdx.x & 63) << 6;   // key slab base

    const float* __restrict__ Vb = V + ((size_t)b * SEQ + s0) * DIM;
#pragma unroll
    for (int i = 0; i < 16; ++i) {
        const int idx = threadIdx.x + (i << 8);  // 0..4095, coalesced in d
        const int s = idx >> 6, d = idx & 63;
        tile[s][d] = (_Float16)Vb[(size_t)s * DIM + d];
    }
    __syncthreads();

    _Float16* __restrict__ Ob = Vt + (size_t)b * DIM * SEQ;
    const int d  = threadIdx.x >> 2;            // 0..63
    const int so = (threadIdx.x & 3) << 4;      // 0,16,32,48
#pragma unroll
    for (int j = 0; j < 16; ++j)
        Ob[(size_t)d * SEQ + s0 + so + j] = tile[so + j][d];
}

// ---------------------------------------------------------------------------
// Main attention kernel: f16 operands pre-staged (K16 row-major, Vt16
// transposed), Q converted in-register once per workgroup.
// ---------------------------------------------------------------------------
__global__ __launch_bounds__(256, 1)
void attn_f16_kernel(const float* __restrict__ Q,
                     const _Float16* __restrict__ K16,
                     const _Float16* __restrict__ Vt16,
                     float* __restrict__ O) {
    __shared__ float s_scores[BATCH][QT][KT];               // 16 KB
    // Attn weights stored directly in WMMA A-operand order: [b][lane][elem].
    __shared__ __align__(32) _Float16 s_attn[BATCH][32][16]; // 8 KB

    const int lane = threadIdx.x & 31;
    const int b    = threadIdx.x >> 5;     // wave == batch
    const int q0   = blockIdx.x * QT;
    const int half = lane >> 4;            // lane half-select (ISA A/B layouts)
    const int lm   = lane & 15;

    const float*    __restrict__ Qb = Q    + ((size_t)b * SEQ + q0) * DIM;
    const _Float16* __restrict__ Kb = K16  + (size_t)b * SEQ * DIM;   // [s][d]
    const _Float16* __restrict__ Vb = Vt16 + (size_t)b * DIM * SEQ;   // [d][s]

    // ---- Preload Q A-tiles: A[16(M=q) x 32(K=d)] f16, two d-chunks.
    // A as loaded here (contiguous k, same order as B): lane -> m = lane%16, halfsel = lane/16;
    // element e -> K = halfsel*16 + e.
    v16h a0, a1;
#pragma unroll
    for (int e = 0; e < 16; ++e) {
        const int d = (half << 4) + e;
        a0[e] = (_Float16)Qb[lm * DIM + d];
        a1[e] = (_Float16)Qb[lm * DIM + d + 32];
    }

    v8f acc[4];
#pragma unroll
    for (int c = 0; c < 4; ++c) acc[c] = (v8f){};

    for (int kc = 0; kc < SEQ; kc += KT) {
        // Prefetch next chunk (global_prefetch_b8): 32 K rows + 64 Vt rows.
        if (kc + KT < SEQ) {
            __builtin_prefetch(&Kb[(size_t)(kc + KT + lane) * DIM], 0, 1);
            __builtin_prefetch(&Vb[(size_t)lane * SEQ + kc + KT], 0, 1);
            __builtin_prefetch(&Vb[(size_t)(lane + 32) * SEQ + kc + KT], 0, 1);
        }

        // ---- GEMM 1: S[16q x 32k] = Q[16x64] . K^T. B tile [32(K=d) x 16(N=key)]:
        // lane -> n = lane%16, element e -> row d = (lane/16)*16 + e.
        // 16 contiguous f16 along d -> one v16h (2x global_load_b128), no cvt.
        const size_t r0 = (size_t)(kc + lm) * DIM + (half << 4);
        const size_t r1 = (size_t)(kc + 16 + lm) * DIM + (half << 4);
        const v16h bk00 = *(const v16h*)(Kb + r0);
        const v16h bk01 = *(const v16h*)(Kb + r0 + 32);
        const v16h bk10 = *(const v16h*)(Kb + r1);
        const v16h bk11 = *(const v16h*)(Kb + r1 + 32);
        SCHED_FENCE();   // keep the 8 b128 loads clustered ahead of the WMMAs

        v8f sc0 = (v8f){}, sc1 = (v8f){};
        sc0 = wmma_f16(a0, bk00, sc0);
        sc0 = wmma_f16(a1, bk01, sc0);
        sc1 = wmma_f16(a0, bk10, sc1);
        sc1 = wmma_f16(a1, bk11, sc1);
        SCHED_FENCE();

        // ---- Issue GEMM-2 V tiles NOW: independent of LDS, their latency is
        // hidden under barrier A + the softmax exp work.
        // B' tile [32(K=key) x 16(N=d)]: lane -> n = lane%16 (d), element e ->
        // row key = (lane/16)*16 + e; contiguous in Vt16[d][s].
        v16h bv[4];
#pragma unroll
        for (int c = 0; c < 4; ++c)
            bv[c] = *(const v16h*)(Vb + (size_t)((c << 4) + lm) * SEQ
                                      + kc + (half << 4));
        SCHED_FENCE();

        // C layout: VGPR r -> q = r + 8*(lane/16), n = lane%16. Scale 1/sqrt(64).
#pragma unroll
        for (int r = 0; r < 8; ++r) {
            s_scores[b][r + (half << 3)][lm]      = sc0[r] * 0.125f;
            s_scores[b][r + (half << 3)][lm + 16] = sc1[r] * 0.125f;
        }
        __syncthreads();   // barrier A: scores visible

        // ---- Softmax across the BATCH axis: 512 (q,k) elems, 2 per thread.
        // Write results directly in A-operand order:
        //   (qq,kk) -> lane = (kk>>4)*16 + qq, elem = kk&15.
        for (int t = threadIdx.x; t < QT * KT; t += 256) {
            const int qq = t >> 5, kk = t & (KT - 1);
            const int wlane = ((kk >> 4) << 4) + qq;
            const int welem = kk & 15;
            float vals[BATCH];
            float m = -3.0e38f;
#pragma unroll
            for (int bb = 0; bb < BATCH; ++bb) {
                vals[bb] = s_scores[bb][qq][kk];
                m = fmaxf(m, vals[bb]);
            }
            float sum = 0.f;
#pragma unroll
            for (int bb = 0; bb < BATCH; ++bb) {
                vals[bb] = __expf(vals[bb] - m);
                sum += vals[bb];
            }
            const float inv = 1.f / sum;
#pragma unroll
            for (int bb = 0; bb < BATCH; ++bb)
                s_attn[bb][wlane][welem] = (_Float16)(vals[bb] * inv);
        }
        __syncthreads();   // barrier B: attn visible

        // ---- Reload attn A'[16x32]: already in A layout -> 2x ds_load_b128.
        const v16h at = *(const v16h*)&s_attn[b][lane][0];

        // ---- GEMM 2: O[16q x 64d] += attn[16x32] . V[32x16] per d-chunk.
        // bv tiles were issued before the softmax; loadcnt should be drained.
#pragma unroll
        for (int c = 0; c < 4; ++c)
            acc[c] = wmma_f16(at, bv[c], acc[c]);
        // No third barrier needed: next iteration's s_scores writes are
        // separated from this softmax's reads by barrier B, and its s_attn
        // writes from this GEMM's reads by the next barrier A.
    }

    // ---- Store outputs: C layout VGPR r -> q = r + 8*half, n = lane%16.
    float* __restrict__ Ob = O + ((size_t)b * SEQ + q0) * DIM;
#pragma unroll
    for (int r = 0; r < 8; ++r) {
        const int qq = r + (half << 3);
#pragma unroll
        for (int c = 0; c < 4; ++c)
            Ob[qq * DIM + (c << 4) + lm] = acc[c][r];
    }
}

// ---------------------------------------------------------------------------
// Fallback (ws too small): direct f32 loads + in-loop conversion.
// ---------------------------------------------------------------------------
__global__ __launch_bounds__(256, 1)
void attn_f32src_kernel(const float* __restrict__ Q,
                        const float* __restrict__ K,
                        const float* __restrict__ V,
                        float* __restrict__ O) {
    __shared__ float    s_scores[BATCH][QT][KT];
    __shared__ _Float16 s_attn  [BATCH][QT][KT];

    const int lane = threadIdx.x & 31;
    const int b    = threadIdx.x >> 5;
    const int q0   = blockIdx.x * QT;
    const int half = lane >> 4;
    const int lm   = lane & 15;

    const float* __restrict__ Qb = Q + ((size_t)b * SEQ + q0) * DIM;
    const float* __restrict__ Kb = K + (size_t)b * SEQ * DIM;
    const float* __restrict__ Vb = V + (size_t)b * SEQ * DIM;

    v16h a0, a1;
#pragma unroll
    for (int e = 0; e < 16; ++e) {
        const int d = (half << 4) + e;
        a0[e] = (_Float16)Qb[lm * DIM + d];
        a1[e] = (_Float16)Qb[lm * DIM + d + 32];
    }

    v8f acc[4];
#pragma unroll
    for (int c = 0; c < 4; ++c) acc[c] = (v8f){};

    for (int kc = 0; kc < SEQ; kc += KT) {
        v16h bk00, bk01, bk10, bk11;
#pragma unroll
        for (int e = 0; e < 16; ++e) {
            const int dr = (half << 4) + e;
            const size_t r0 = (size_t)(kc + lm) * DIM;
            const size_t r1 = (size_t)(kc + 16 + lm) * DIM;
            bk00[e] = (_Float16)Kb[r0 + dr];
            bk01[e] = (_Float16)Kb[r0 + dr + 32];
            bk10[e] = (_Float16)Kb[r1 + dr];
            bk11[e] = (_Float16)Kb[r1 + dr + 32];
        }
        v8f sc0 = (v8f){}, sc1 = (v8f){};
        sc0 = wmma_f16(a0, bk00, sc0);
        sc0 = wmma_f16(a1, bk01, sc0);
        sc1 = wmma_f16(a0, bk10, sc1);
        sc1 = wmma_f16(a1, bk11, sc1);

#pragma unroll
        for (int r = 0; r < 8; ++r) {
            s_scores[b][r + (half << 3)][lm]      = sc0[r] * 0.125f;
            s_scores[b][r + (half << 3)][lm + 16] = sc1[r] * 0.125f;
        }
        __syncthreads();

        for (int t = threadIdx.x; t < QT * KT; t += 256) {
            const int qq = t >> 5, kk = t & (KT - 1);
            float vals[BATCH];
            float m = -3.0e38f;
#pragma unroll
            for (int bb = 0; bb < BATCH; ++bb) {
                vals[bb] = s_scores[bb][qq][kk];
                m = fmaxf(m, vals[bb]);
            }
            float sum = 0.f;
#pragma unroll
            for (int bb = 0; bb < BATCH; ++bb) {
                vals[bb] = __expf(vals[bb] - m);
                sum += vals[bb];
            }
            const float inv = 1.f / sum;
#pragma unroll
            for (int bb = 0; bb < BATCH; ++bb)
                s_attn[bb][qq][kk] = (_Float16)(vals[bb] * inv);
        }
        __syncthreads();

        v16h at;
#pragma unroll
        for (int e = 0; e < 16; ++e) {
            const int kk = (half << 4) + e;
            at[e] = s_attn[b][lm][kk];
        }
        __syncthreads();

#pragma unroll
        for (int c = 0; c < 4; ++c) {
            v16h bv;
#pragma unroll
            for (int e = 0; e < 16; ++e)
                bv[e] = (_Float16)Vb[(size_t)(kc + (half << 4) + e) * DIM
                                     + (c << 4) + lm];
            acc[c] = wmma_f16(at, bv, acc[c]);
        }
    }

    float* __restrict__ Ob = O + ((size_t)b * SEQ + q0) * DIM;
#pragma unroll
    for (int r = 0; r < 8; ++r) {
        const int qq = r + (half << 3);
#pragma unroll
        for (int c = 0; c < 4; ++c)
            Ob[qq * DIM + (c << 4) + lm] = acc[c][r];
    }
}

extern "C" void kernel_launch(void* const* d_in, const int* in_sizes, int n_in,
                              void* d_out, int out_size, void* d_ws, size_t ws_size,
                              hipStream_t stream) {
    (void)in_sizes; (void)n_in; (void)out_size;
    const float* q = (const float*)d_in[0];
    const float* k = (const float*)d_in[1];
    const float* v = (const float*)d_in[2];
    float* o = (float*)d_out;

    const size_t nElem    = (size_t)BATCH * SEQ * DIM;   // 2,097,152
    const size_t f16Bytes = nElem * sizeof(_Float16);    // 4 MiB each

    if (ws_size >= 2 * f16Bytes) {
        _Float16* k16  = (_Float16*)d_ws;
        _Float16* vt16 = (_Float16*)((char*)d_ws + f16Bytes);

        convert_k_kernel<<<dim3(nElem / (256 * 4)), dim3(256), 0, stream>>>(k, k16);
        transpose_v_kernel<<<dim3(BATCH * (SEQ / 64)), dim3(256), 0, stream>>>(v, vt16);
        attn_f16_kernel<<<dim3(SEQ / QT), dim3(256), 0, stream>>>(q, k16, vt16, o);
    } else {
        attn_f32src_kernel<<<dim3(SEQ / QT), dim3(256), 0, stream>>>(q, k, v, o);
    }
}
